// NonLocalBlock2D_36661840839099
// MI455X (gfx1250) — matrix operand order repaired and checked by reading, not compiled, not run on hardware
//
#include <hip/hip_runtime.h>
#include <hip/hip_bf16.h>

// ---------------------------------------------------------------------------
// NonLocalBlock2D for MI455X (gfx1250, wave32, WMMA).
// Flash-attention style: f (4 x 4096 x 4096) is never materialized in HBM.
//   pass0: fused unshuffle + 3 projections -> th[Np,32]f16, phT[Np,32]f16,
//          g[32,Np]f16 (per batch)
//   pass1: column softmax stats (online max/sum) via v_wmma_f32_16x16x32_f16
//   pass2: recompute f^T tiles, exp-normalize in-register, feed as WMMA B
//          operand (shfl_xor(16) half-swap, no LDS transpose), accumulate
//          y^T = g * P^T with WMMA, write y[32,Np]f32
//   pass3: W projection [496x32] + pixel-shuffle scatter (VALU, trivial)
// ---------------------------------------------------------------------------

typedef __attribute__((ext_vector_type(16))) _Float16 v16h;
typedef __attribute__((ext_vector_type(8)))  _Float16 v8h;
typedef __attribute__((ext_vector_type(8)))  float    v8f;

#define NB   4          // batch
#define CCH  31         // num_wavelength
#define SC   4          // scale
#define HH   256        // H = W
#define HS   64         // h = w = H/SC
#define NP   4096       // h*w
#define ICH  32         // inter channels
#define CINU 496        // SC*SC*CCH

// ---- WMMA operand loaders (contiguous k order, same in A and B) -------
// A 16x32 f16: lane m holds row M=m; hi=0 -> K 0..15, hi=1 -> K 16..31 (contiguous)
__device__ __forceinline__ v16h load_tileA(const _Float16* __restrict__ row, int hi) {
    v8h lo = *(const v8h*)(row + hi * 16);
    v8h hh = *(const v8h*)(row + hi * 16 + 8);
    return __builtin_shufflevector(lo, hh, 0,1,2,3,4,5,6,7,8,9,10,11,12,13,14,15);
}
// B 32x16 f16: lane m holds col N=m; hi=0 -> K 0..15, hi=1 -> K 16..31 (contiguous)
__device__ __forceinline__ v16h load_tileB(const _Float16* __restrict__ row, int hi) {
    return *(const v16h*)(row + hi * 16);
}

// ---- pass 0: unshuffle + 1x1 conv projections ------------------------------
// grid (NP/8, NB), block 256.  lane = out-channel o, (t>>5) = local pixel.
__global__ void nlb_proj(const float* __restrict__ x,
                         const float* __restrict__ gw, const float* __restrict__ gb,
                         const float* __restrict__ tw, const float* __restrict__ tb,
                         const float* __restrict__ pw, const float* __restrict__ pb,
                         _Float16* __restrict__ th, _Float16* __restrict__ phT,
                         _Float16* __restrict__ g) {
    const int t  = threadIdx.x;
    const int o  = t & 31;
    const int p  = blockIdx.x * 8 + (t >> 5);
    const int n  = blockIdx.y;
    const int i  = p >> 6;          // row in 64
    const int j  = p & 63;          // col in 64
    const float* xn = x + (size_t)n * CCH * HH * HH;

    float ag = gb[o], at = tb[o], ap = pb[o];
    #pragma unroll
    for (int sh = 0; sh < SC; ++sh) {
        #pragma unroll
        for (int sw = 0; sw < SC; ++sw) {
            const int cu0 = (sh * SC + sw) * CCH;
            const float* xb = xn + (size_t)(i * SC + sh) * HH + (j * SC + sw);
            const float* gwr = gw + (size_t)o * CINU + cu0;
            const float* twr = tw + (size_t)o * CINU + cu0;
            const float* pwr = pw + (size_t)o * CINU + cu0;
            for (int c = 0; c < CCH; ++c) {
                float xv = xb[(size_t)c * (HH * HH)];   // uniform across wave
                ag = fmaf(gwr[c], xv, ag);
                at = fmaf(twr[c], xv, at);
                ap = fmaf(pwr[c], xv, ap);
            }
        }
    }
    const size_t rb = ((size_t)n * NP + p) * ICH + o;
    th[rb]  = (_Float16)at;
    phT[rb] = (_Float16)ap;
    g[((size_t)n * ICH + o) * NP + p] = (_Float16)ag;   // [32, Np] per batch
}

// ---- pass 1: per-column online softmax stats -------------------------------
// One wave per (n, 16-column block). f tile: A = th rows (M=i), B = phT rows (N=j).
// D layout: lane holds column j = j0 + (lane&15), rows i = r + 8*(lane>>4).
__global__ void nlb_colstats(const _Float16* __restrict__ th,
                             const _Float16* __restrict__ phT,
                             float* __restrict__ colM, float* __restrict__ colZ) {
    const int lane = threadIdx.x & 31;
    const int gidx = blockIdx.x * 8 + (threadIdx.x >> 5);  // 0..1023
    const int n  = gidx >> 8;
    const int j0 = (gidx & 255) * 16;
    const int m  = lane & 15, hi = lane >> 4;
    const _Float16* thn = th  + (size_t)n * NP * ICH;
    const _Float16* phn = phT + (size_t)n * NP * ICH;

    const v16h B = load_tileB(phn + (size_t)(j0 + m) * ICH, hi);

    float mx = -1e30f, sm = 0.0f;
    for (int i0 = 0; i0 < NP; i0 += 16) {
        const v16h A = load_tileA(thn + (size_t)(i0 + m) * ICH, hi);
        v8f acc = {};
        acc = __builtin_amdgcn_wmma_f32_16x16x32_f16(false, A, false, B,
                                                     (short)0, acc, false, false);
        float tm = mx;
        #pragma unroll
        for (int r = 0; r < 8; ++r) tm = fmaxf(tm, acc[r]);
        float s2 = 0.0f;
        #pragma unroll
        for (int r = 0; r < 8; ++r) s2 += __expf(acc[r] - tm);
        sm = sm * __expf(mx - tm) + s2;
        mx = tm;
    }
    // combine the two lane-halves holding the same column
    const float mo = __shfl_xor(mx, 16, 32);
    const float so = __shfl_xor(sm, 16, 32);
    const float Mf = fmaxf(mx, mo);
    const float Zf = sm * __expf(mx - Mf) + so * __expf(mo - Mf);
    if (hi == 0) {
        colM[(size_t)n * NP + j0 + m] = Mf;
        colZ[(size_t)n * NP + j0 + m] = 1.0f / Zf;
    }
}

// ---- pass 2: y^T = g * softmax(f)^T, streaming over j ----------------------
// One wave per (n, 16-row block i0). fT tile: A = phT rows (M=j), B = th rows (N=i).
// D(fT) lane holds column i, rows j = r+8*hi -> after a lane<->lane+16 swap this
// is exactly the B-operand layout (lane=N=i, K=j) for the second WMMA.
__global__ void nlb_attn(const _Float16* __restrict__ th,
                         const _Float16* __restrict__ phT,
                         const _Float16* __restrict__ g,
                         const float* __restrict__ colM, const float* __restrict__ colZ,
                         float* __restrict__ y) {
    const int lane = threadIdx.x & 31;
    const int gidx = blockIdx.x * 8 + (threadIdx.x >> 5);  // 0..1023
    const int n  = gidx >> 8;
    const int i0 = (gidx & 255) * 16;
    const int m  = lane & 15, hi = lane >> 4;
    const _Float16* thn = th  + (size_t)n * NP * ICH;
    const _Float16* phn = phT + (size_t)n * NP * ICH;
    const _Float16* gn  = g   + (size_t)n * ICH * NP;
    const float* Mn = colM + (size_t)n * NP;
    const float* Zn = colZ + (size_t)n * NP;

    const v16h Bth = load_tileB(thn + (size_t)(i0 + m) * ICH, hi);  // fixed B (N=i)

    v8f y0 = {}, y1 = {};
    for (int jc = 0; jc < NP; jc += 32) {
        // two fT tiles covering j = jc..jc+31
        const v16h A0 = load_tileA(phn + (size_t)(jc + m) * ICH, hi);
        const v16h A1 = load_tileA(phn + (size_t)(jc + 16 + m) * ICH, hi);
        v8f f0 = {}, f1 = {};
        f0 = __builtin_amdgcn_wmma_f32_16x16x32_f16(false, A0, false, Bth,
                                                    (short)0, f0, false, false);
        f1 = __builtin_amdgcn_wmma_f32_16x16x32_f16(false, A1, false, Bth,
                                                    (short)0, f1, false, false);
        // softmax-normalize: this lane holds rows j = jc + 8*hi + r (tile0), +16 (tile1)
        float p0[8], p1[8];
        #pragma unroll
        for (int r = 0; r < 8; ++r) {
            const int ja = jc + 8 * hi + r;
            const int jb = ja + 16;
            p0[r] = __expf(f0[r] - Mn[ja]) * Zn[ja];
            p1[r] = __expf(f1[r] - Mn[jb]) * Zn[jb];
        }
        // half-swap so each lane owns a contiguous K range (B layout):
        // lane hi=0 needs K 0..15 = own p0 + partner p0; hi=1 needs K 16..31.
        v16h P = {};
        #pragma unroll
        for (int r = 0; r < 8; ++r) {
            const float send = hi ? p0[r] : p1[r];
            const float recv = __shfl_xor(send, 16, 32);
            P[r]     = (_Float16)(hi ? recv : p0[r]);   // K = hi*16 + r
            P[8 + r] = (_Float16)(hi ? p1[r] : recv);   // K = hi*16 + 8 + r
        }
        // y^T += g(rows c) x P : A = g rows, K along j (contiguous in g[32,Np])
        const v16h GA0 = load_tileA(gn + (size_t)m * NP + jc, hi);
        const v16h GA1 = load_tileA(gn + (size_t)(16 + m) * NP + jc, hi);
        y0 = __builtin_amdgcn_wmma_f32_16x16x32_f16(false, GA0, false, P,
                                                    (short)0, y0, false, false);
        y1 = __builtin_amdgcn_wmma_f32_16x16x32_f16(false, GA1, false, P,
                                                    (short)0, y1, false, false);
    }
    // D layout: lane holds column i = i0+m, rows c = r+8*hi (+16 for y1)
    float* yn = y + (size_t)n * ICH * NP;
    #pragma unroll
    for (int r = 0; r < 8; ++r) {
        yn[(size_t)(r + 8 * hi) * NP + i0 + m]      = y0[r];
        yn[(size_t)(16 + r + 8 * hi) * NP + i0 + m] = y1[r];
    }
}

// ---- pass 3: W projection [496x32] + pixel shuffle back to [N,31,256,256] --
__global__ void nlb_wproj(const float* __restrict__ y,
                          const float* __restrict__ Ww, const float* __restrict__ Wb,
                          float* __restrict__ z) {
    const long idx = (long)blockIdx.x * blockDim.x + threadIdx.x;  // N*496*NP threads
    const int p = (int)(idx & (NP - 1));
    const int u = (int)((idx >> 12) % CINU);
    const int n = (int)(idx / ((long)NP * CINU));
    const float* yn = y + (size_t)n * ICH * NP + p;
    const float* wr = Ww + (size_t)u * ICH;
    float acc = Wb[u];
    #pragma unroll
    for (int c = 0; c < ICH; ++c) acc = fmaf(wr[c], yn[(size_t)c * NP], acc);
    const int cch = u % CCH;
    const int ss  = u / CCH;           // sh*4 + sw
    const int sw  = ss & 3, sh = ss >> 2;
    const int i = p >> 6, j = p & 63;
    z[((size_t)(n * CCH + cch) * HH + (i * SC + sh)) * HH + (j * SC + sw)] = acc;
}

extern "C" void kernel_launch(void* const* d_in, const int* in_sizes, int n_in,
                              void* d_out, int out_size, void* d_ws, size_t ws_size,
                              hipStream_t stream) {
    (void)in_sizes; (void)n_in; (void)out_size; (void)ws_size;
    const float* x  = (const float*)d_in[0];
    const float* gw = (const float*)d_in[1];
    const float* gb = (const float*)d_in[2];
    const float* tw = (const float*)d_in[3];
    const float* tb = (const float*)d_in[4];
    const float* pw = (const float*)d_in[5];
    const float* pb = (const float*)d_in[6];
    const float* Ww = (const float*)d_in[7];
    const float* Wb = (const float*)d_in[8];
    float* z = (float*)d_out;

    char* ws = (char*)d_ws;                       // ~5.2 MB used
    _Float16* th  = (_Float16*)(ws);                              // 1 MB
    _Float16* phT = (_Float16*)(ws + (1u << 20));                 // 1 MB
    _Float16* g   = (_Float16*)(ws + (2u << 20));                 // 1 MB
    float*    cM  = (float*)(ws + (3u << 20));                    // 64 KB
    float*    cZ  = (float*)(ws + (3u << 20) + (1u << 16));       // 64 KB
    float*    y   = (float*)(ws + (3u << 20) + (2u << 16));       // 2 MB

    nlb_proj<<<dim3(NP / 8, NB), 256, 0, stream>>>(x, gw, gb, tw, tb, pw, pb,
                                                   th, phT, g);
    nlb_colstats<<<128, 256, 0, stream>>>(th, phT, cM, cZ);
    nlb_attn<<<128, 256, 0, stream>>>(th, phT, g, cM, cZ, y);
    nlb_wproj<<<(NB * CINU * NP) / 256, 256, 0, stream>>>(y, Ww, Wb, z);
}
